// Model_14809047236615
// MI455X (gfx1250) — matrix operand order repaired and checked by reading, not compiled, not run on hardware
//
#include <hip/hip_runtime.h>
#include <hip/hip_bf16.h>
#include <math.h>

// ---------------- CDNA5 WMMA types ----------------
typedef __attribute__((ext_vector_type(16))) _Float16 v16h;
typedef __attribute__((ext_vector_type(8)))  _Float16 h8;
typedef __attribute__((ext_vector_type(8)))  float    v8f;

#define HID 128
#define NLAYERS 4

__device__ __forceinline__ float lrelu(float v) { return v > 0.f ? v : 0.01f * v; }
__device__ __forceinline__ float sigmoidf(float v) { return 1.f / (1.f + __expf(-v)); }

// order-preserving float->uint map for atomic max on signed floats
__device__ __forceinline__ unsigned fenc(float f) {
    unsigned u = __float_as_uint(f);
    return (u & 0x80000000u) ? ~u : (u | 0x80000000u);
}
__device__ __forceinline__ float fdec(unsigned u) {
    unsigned v = (u & 0x80000000u) ? (u & 0x7FFFFFFFu) : ~u;
    return __uint_as_float(v);
}

// ---------------- weight pre-pass: W[K,Nc] f32 -> Wt[Nc,K] f16 (transposed) ----------------
__global__ __launch_bounds__(256) void wt_convert_kernel(
    const float* __restrict__ W, _Float16* __restrict__ Wt, int K, int Nc) {
    int total = K * Nc;
    int stride = gridDim.x * blockDim.x;
    for (int idx = blockIdx.x * blockDim.x + threadIdx.x; idx < total; idx += stride) {
        int n = idx / K, k = idx - n * K;          // write-coalesced over idx
        Wt[(long)n * K + k] = (_Float16)W[(long)k * Nc + n];
    }
}

// ---------------- node encoder: x[N,16] @ Wn[16,128] + bn ----------------
__global__ __launch_bounds__(128) void node_enc_kernel(
    const float* __restrict__ x, const float* __restrict__ Wn,
    const float* __restrict__ bn, float* __restrict__ out, int N) {
    int row = blockIdx.x;
    int c = threadIdx.x;
    if (row >= N) return;
    __shared__ float xr[16];
    if (c < 16) xr[c] = x[row * 16 + c];
    __syncthreads();
    float s = bn[c];
#pragma unroll
    for (int k = 0; k < 16; ++k) s += xr[k] * Wn[k * HID + c];
    out[(long)row * HID + c] = s;
}

// ---------------- per-row attention dots al/ar ----------------
__global__ __launch_bounds__(256) void alar_kernel(
    const float* __restrict__ x, const float* __restrict__ attl,
    const float* __restrict__ attr, float* __restrict__ al,
    float* __restrict__ ar, int N) {
    int wv = threadIdx.x >> 5, lane = threadIdx.x & 31;
    int row = blockIdx.x * 8 + wv;
    if (row >= N) return;
    float sl = 0.f, sr = 0.f;
#pragma unroll
    for (int k = lane; k < HID; k += 32) {
        float v = x[(long)row * HID + k];
        sl += v * attl[k];
        sr += v * attr[k];
    }
#pragma unroll
    for (int off = 16; off; off >>= 1) {
        sl += __shfl_xor(sl, off, 32);
        sr += __shfl_xor(sr, off, 32);
    }
    if (lane == 0) { al[row] = sl; ar[row] = sr; }
}

// ---------------- per-layer init: zero agg, reset amax/denom/BN sums ----------------
__global__ __launch_bounds__(256) void layer_init_kernel(
    float4* __restrict__ agg4, unsigned* __restrict__ amax,
    float* __restrict__ denom, float* __restrict__ sums, int N) {
    int i = blockIdx.x * blockDim.x + threadIdx.x;
    int stride = gridDim.x * blockDim.x;
    int n4 = N * (HID / 4);
    for (int j = i; j < n4; j += stride) agg4[j] = make_float4(0.f, 0.f, 0.f, 0.f);
    for (int j = i; j < N; j += stride) { amax[j] = fenc(-3.0e38f); denom[j] = 0.f; }
    if (i < 512) sums[i] = 0.f;
}

// ---------------- edge logits + segment max ----------------
__global__ __launch_bounds__(256) void logit_kernel(
    const int* __restrict__ ei, const float* __restrict__ al,
    const float* __restrict__ ar, float* __restrict__ exb,
    unsigned* __restrict__ amax, int E) {
    int e = blockIdx.x * blockDim.x + threadIdx.x;
    if (e >= E) return;
    int s = ei[e], d = ei[E + e];
    float a = lrelu(ar[d] + al[s]);
    exb[e] = a;
    atomicMax(&amax[d], fenc(a));
}

// ---------------- exp + segment sum ----------------
__global__ __launch_bounds__(256) void exp_kernel(
    const int* __restrict__ ei, float* __restrict__ exb,
    const unsigned* __restrict__ amax, float* __restrict__ denom, int E) {
    int e = blockIdx.x * blockDim.x + threadIdx.x;
    if (e >= E) return;
    int d = ei[E + e];
    float v = __expf(exb[e] - fdec(amax[d]));
    exb[e] = v;
    atomicAdd(&denom[d], v);
}

// ---------------- message + scatter aggregate (fused edge encoder) ----------------
__global__ __launch_bounds__(256) void message_kernel(
    const float* __restrict__ x, const float* __restrict__ edge_attr,
    const int* __restrict__ ei, const float* __restrict__ We,
    const float* __restrict__ be, const float* __restrict__ exb,
    const float* __restrict__ denom, float* __restrict__ agg, int E) {
    int wv = threadIdx.x >> 5, lane = threadIdx.x & 31;
    long e = (long)blockIdx.x * 8 + wv;
    if (e >= E) return;
    int s = ei[e], d = ei[E + e];
    float alpha = exb[e] / denom[d];
    float ea[8];
#pragma unroll
    for (int r = 0; r < 8; ++r) ea[r] = edge_attr[e * 8 + r];
    int k4 = lane * 4;
    float4 xs = *reinterpret_cast<const float4*>(x + (long)s * HID + k4);
    float4 ev = *reinterpret_cast<const float4*>(be + k4);
#pragma unroll
    for (int r = 0; r < 8; ++r) {
        float4 w = *reinterpret_cast<const float4*>(We + r * HID + k4);
        ev.x += ea[r] * w.x; ev.y += ea[r] * w.y;
        ev.z += ea[r] * w.z; ev.w += ea[r] * w.w;
    }
    float4 m;
    m.x = lrelu(xs.x + ev.x) * alpha;
    m.y = lrelu(xs.y + ev.y) * alpha;
    m.z = lrelu(xs.z + ev.z) * alpha;
    m.w = lrelu(xs.w + ev.w) * alpha;
    float* dp = agg + (long)d * HID + k4;
    atomicAdd(dp + 0, m.x); atomicAdd(dp + 1, m.y);
    atomicAdd(dp + 2, m.z); atomicAdd(dp + 3, m.w);
}

// ---------------- WMMA GEMM: C[M,NDIM] = act(A)[M,KDIM] @ B + bias ----------------
// B pre-converted to f16, K-transposed layout Bt_g[NDIM][KDIM].
// 8 waves per block: MW x NW wave grid, each wave owns 16 rows x 64 cols (4 WMMA n-tiles).
// A staged f32->f16 in LDS (optionally BN scale/shift + LeakyReLU fused into the stage).
template <int KDIM, int NDIM, int MW, int NW, bool APPLY_BN>
__global__ __launch_bounds__(256) void gemm_wmma_kernel(
    const float* __restrict__ A, const _Float16* __restrict__ Bt_g,
    const float* __restrict__ bias, const float* __restrict__ scale,
    const float* __restrict__ shift, float* __restrict__ C, int M) {
    static_assert(MW * NW == 8, "8 waves per block");
    static_assert(NW * 64 == NDIM, "block covers full N");
    constexpr int BM = 16 * MW;
    __shared__ _Float16 As[BM][32];
    __shared__ _Float16 Bt[NDIM][32];

    int tid = threadIdx.x;
    int w = tid >> 5, lane = tid & 31;
    int wm = w / NW, wn = w % NW;
    int half = lane >> 4, l16 = lane & 15;
    int m0 = blockIdx.x * BM;

    v8f acc[4] = {v8f{}, v8f{}, v8f{}, v8f{}};

    for (int kc = 0; kc < KDIM; kc += 32) {
        // stage A tile (BM x 32) as f16, with optional fused BN + LeakyReLU
        for (int idx = tid * 4; idx < BM * 32; idx += 256 * 4) {
            int r = idx >> 5, c = idx & 31;
            int gr = m0 + r;
            float4 v = make_float4(0.f, 0.f, 0.f, 0.f);
            if (gr < M) {
                v = *reinterpret_cast<const float4*>(A + (long)gr * KDIM + kc + c);
                if constexpr (APPLY_BN) {
                    int kk = kc + c;
                    v.x = lrelu(v.x * scale[kk + 0] + shift[kk + 0]);
                    v.y = lrelu(v.y * scale[kk + 1] + shift[kk + 1]);
                    v.z = lrelu(v.z * scale[kk + 2] + shift[kk + 2]);
                    v.w = lrelu(v.w * scale[kk + 3] + shift[kk + 3]);
                }
            }
            As[r][c + 0] = (_Float16)v.x;
            As[r][c + 1] = (_Float16)v.y;
            As[r][c + 2] = (_Float16)v.z;
            As[r][c + 3] = (_Float16)v.w;
        }
        // stage B tile: contiguous 16B copies from pre-transposed f16 weights
        for (int idx = tid * 8; idx < NDIM * 32; idx += 256 * 8) {
            int n = idx >> 5, c = idx & 31;
            *reinterpret_cast<h8*>(&Bt[n][c]) =
                *reinterpret_cast<const h8*>(Bt_g + (long)n * KDIM + kc + c);
        }
        __syncthreads();

        // A fragment: lanes 0-15 K=0..7,16..23 ; lanes 16-31 K=8..15,24..31
        v16h af;
        {
            const _Float16* a0 = &As[wm * 16 + l16][half * 8];
            const _Float16* a1 = &As[wm * 16 + l16][16 + half * 8];
#pragma unroll
            for (int i = 0; i < 8; ++i) { af[i] = a0[i]; af[8 + i] = a1[i]; }
        }
#pragma unroll
        for (int nn = 0; nn < 4; ++nn) {
            int col = wn * 64 + nn * 16 + l16;
            v16h bf;
            const _Float16* bp = &Bt[col][half * 8];  // K=8h..8h+7, 16+8h..16+8h+7 (matches A)
#pragma unroll
            for (int i = 0; i < 8; ++i) { bf[i] = bp[i]; bf[8 + i] = bp[16 + i]; }
            acc[nn] = __builtin_amdgcn_wmma_f32_16x16x32_f16(
                false, af, false, bf, (short)0, acc[nn], false, false);
        }
        __syncthreads();
    }

    // epilogue: +bias, guarded store (C/D layout: VGPR r -> M = r + 8*half, N = lane&15)
#pragma unroll
    for (int nn = 0; nn < 4; ++nn) {
        int col = wn * 64 + nn * 16 + l16;
        float bv = bias[col];
#pragma unroll
        for (int r = 0; r < 8; ++r) {
            int row = m0 + wm * 16 + half * 8 + r;
            if (row < M) C[(long)row * NDIM + col] = acc[nn][r] + bv;
        }
    }
}

// ---------------- BN batch statistics (partial sums per column) ----------------
__global__ __launch_bounds__(256) void bn_stats_kernel(
    const float* __restrict__ H, float* __restrict__ sums, int M) {
    int c = threadIdx.x;  // 256 columns, coalesced across threads
    float s = 0.f, ss = 0.f;
    for (int r = blockIdx.x; r < M; r += gridDim.x) {
        float v = H[(long)r * 256 + c];
        s += v; ss += v * v;
    }
    atomicAdd(&sums[c], s);
    atomicAdd(&sums[256 + c], ss);
}

__global__ __launch_bounds__(256) void bn_finalize_kernel(
    const float* __restrict__ sums, const float* __restrict__ gamma,
    const float* __restrict__ beta, float* __restrict__ scale,
    float* __restrict__ shift, int M) {
    int c = threadIdx.x;
    float inv = 1.f / (float)M;
    float mu = sums[c] * inv;
    float var = sums[256 + c] * inv - mu * mu;
    float rs = rsqrtf(var + 1e-5f);
    float sc = rs * gamma[c];
    scale[c] = sc;
    shift[c] = beta[c] - mu * sc;
}

// ---------------- shared LSTM cell over virtual nodes ----------------
__global__ __launch_bounds__(512) void lstm_kernel(
    const float* __restrict__ xmlp, const int* __restrict__ v_idx,
    const float* __restrict__ W_ih, const float* __restrict__ W_hh,
    const float* __restrict__ b_ih, const float* __restrict__ b_hh,
    const float* __restrict__ identity, float* __restrict__ hx,
    float* __restrict__ cx, int init) {
    __shared__ float xv[HID], hv[HID], gates[4 * HID];
    int v = blockIdx.x;
    int t = threadIdx.x;
    int node = v_idx[v];
    if (t < HID) {
        xv[t] = xmlp[(long)node * HID + t];
        hv[t] = init ? identity[(long)node * HID + t] : hx[v * HID + t];
    }
    __syncthreads();
    float g = b_ih[t] + b_hh[t];
#pragma unroll 8
    for (int k = 0; k < HID; ++k)
        g += xv[k] * W_ih[k * 512 + t] + hv[k] * W_hh[k * 512 + t];
    gates[t] = g;
    __syncthreads();
    if (t < HID) {
        float ig = sigmoidf(gates[t]);
        float fg = sigmoidf(gates[HID + t]);
        float gg = tanhf(gates[2 * HID + t]);
        float og = sigmoidf(gates[3 * HID + t]);
        float c0 = init ? 0.f : cx[v * HID + t];
        float c1 = fg * c0 + ig * gg;
        cx[v * HID + t] = c1;
        hx[v * HID + t] = og * tanhf(c1);
    }
}

// ---------------- residual add ----------------
__global__ __launch_bounds__(256) void resid_kernel(
    float4* __restrict__ xd, const float4* __restrict__ add, int n4) {
    int i = blockIdx.x * blockDim.x + threadIdx.x;
    int stride = gridDim.x * blockDim.x;
    for (int j = i; j < n4; j += stride) {
        float4 a = xd[j], b = add[j];
        a.x += b.x; a.y += b.y; a.z += b.z; a.w += b.w;
        xd[j] = a;
    }
}

// ---------------- output head ----------------
__global__ __launch_bounds__(256) void out_kernel(
    const float* __restrict__ hx, const float* __restrict__ Wout,
    const float* __restrict__ bout, float* __restrict__ out, int G) {
    int wv = threadIdx.x >> 5, lane = threadIdx.x & 31;
    int row = blockIdx.x * 8 + wv;
    if (row >= G) return;
    float s = 0.f;
#pragma unroll
    for (int k = lane; k < HID; k += 32) s += hx[row * HID + k] * Wout[k];
#pragma unroll
    for (int off = 16; off; off >>= 1) s += __shfl_xor(s, off, 32);
    if (lane == 0) out[row] = sigmoidf(s + bout[0]);
}

// ====================================================================
extern "C" void kernel_launch(void* const* d_in, const int* in_sizes, int n_in,
                              void* d_out, int out_size, void* d_ws, size_t ws_size,
                              hipStream_t stream) {
    const float* x_in      = (const float*)d_in[0];
    const float* edge_attr = (const float*)d_in[1];
    const int*   ei        = (const int*)d_in[2];
    const int*   v_idx     = (const int*)d_in[3];
    const float* Wn   = (const float*)d_in[4];
    const float* bn   = (const float*)d_in[5];
    const float* We   = (const float*)d_in[6];
    const float* be   = (const float*)d_in[7];
    const float* attL = (const float*)d_in[8];
    const float* attR = (const float*)d_in[9];
    const float* W1   = (const float*)d_in[10];
    const float* b1   = (const float*)d_in[11];
    const float* gam  = (const float*)d_in[12];
    const float* bet  = (const float*)d_in[13];
    const float* W2   = (const float*)d_in[14];
    const float* b2   = (const float*)d_in[15];
    const float* W_ih = (const float*)d_in[16];
    const float* W_hh = (const float*)d_in[17];
    const float* b_ih = (const float*)d_in[18];
    const float* b_hh = (const float*)d_in[19];
    const float* Wout = (const float*)d_in[20];
    const float* bout = (const float*)d_in[21];

    const int N = in_sizes[0] / 16;
    const int E = in_sizes[1] / 8;
    const int G = in_sizes[3];

    // workspace carve-up (f32 slots)
    float* w = (float*)d_ws;
    size_t o = 0;
    float*    bufX   = w + o; o += (size_t)N * HID;   // current x / identity
    float*    bufAgg = w + o; o += (size_t)N * HID;   // scatter-add aggregate
    float*    bufMid = w + o; o += (size_t)N * 256;   // hmid
    float*    bufOut = w + o; o += (size_t)N * HID;   // post-MLP (pre-residual)
    float*    exb    = w + o; o += (size_t)E;         // logits -> exp
    float*    al     = w + o; o += (size_t)N;
    float*    ar     = w + o; o += (size_t)N;
    unsigned* amaxU  = (unsigned*)(w + o); o += (size_t)N;
    float*    denom  = w + o; o += (size_t)N;
    float*    sums   = w + o; o += 512;               // [sum | sumsq]
    float*    scale  = w + o; o += 256;
    float*    shift  = w + o; o += 256;
    float*    hx     = w + o; o += (size_t)G * HID;
    float*    cx     = w + o; o += (size_t)G * HID;
    // f16 pre-transposed weights: W1t[l][256][128], W2t[l][128][256]
    _Float16* W1t = (_Float16*)(w + o); o += (size_t)NLAYERS * 128 * 256 / 2;
    _Float16* W2t = (_Float16*)(w + o); o += (size_t)NLAYERS * 256 * 128 / 2;
    (void)ws_size; (void)n_in; (void)out_size;

    // one-time weight conversion: f32 [K,N] -> f16 transposed [N,K]
    for (int l = 0; l < NLAYERS; ++l) {
        wt_convert_kernel<<<128, 256, 0, stream>>>(
            W1 + (size_t)l * 128 * 256, W1t + (size_t)l * 128 * 256, 128, 256);
        wt_convert_kernel<<<128, 256, 0, stream>>>(
            W2 + (size_t)l * 256 * 128, W2t + (size_t)l * 256 * 128, 256, 128);
    }

    // node embedding (edge embedding is fused into message_kernel)
    node_enc_kernel<<<N, 128, 0, stream>>>(x_in, Wn, bn, bufX, N);

    for (int l = 0; l < NLAYERS; ++l) {
        alar_kernel<<<(N + 7) / 8, 256, 0, stream>>>(
            bufX, attL + l * HID, attR + l * HID, al, ar, N);
        layer_init_kernel<<<2048, 256, 0, stream>>>(
            (float4*)bufAgg, amaxU, denom, sums, N);
        logit_kernel<<<(E + 255) / 256, 256, 0, stream>>>(ei, al, ar, exb, amaxU, E);
        exp_kernel<<<(E + 255) / 256, 256, 0, stream>>>(ei, exb, amaxU, denom, E);
        message_kernel<<<(E + 7) / 8, 256, 0, stream>>>(
            bufX, edge_attr, ei, We, be, exb, denom, bufAgg, E);

        // GEMM1: [N,128] @ W1[128,256] + b1 -> bufMid
        gemm_wmma_kernel<128, 256, 2, 4, false><<<(N + 31) / 32, 256, 0, stream>>>(
            bufAgg, W1t + (size_t)l * 128 * 256, b1 + l * 256,
            nullptr, nullptr, bufMid, N);

        bn_stats_kernel<<<512, 256, 0, stream>>>(bufMid, sums, N);
        bn_finalize_kernel<<<1, 256, 0, stream>>>(
            sums, gam + l * 256, bet + l * 256, scale, shift, N);

        // GEMM2: lrelu(BN(bufMid)) @ W2[256,128] + b2 -> bufOut (BN+act fused into A-stage)
        gemm_wmma_kernel<256, 128, 4, 2, true><<<(N + 63) / 64, 256, 0, stream>>>(
            bufMid, W2t + (size_t)l * 256 * 128, b2 + l * 128,
            scale, shift, bufOut, N);

        // LSTM over virtual nodes (reads pre-residual bufOut; bufX still holds identity)
        lstm_kernel<<<G, 512, 0, stream>>>(
            bufOut, v_idx, W_ih, W_hh, b_ih, b_hh, bufX, hx, cx, l == 0 ? 1 : 0);

        // residual: x = postMLP + identity
        resid_kernel<<<2048, 256, 0, stream>>>(
            (float4*)bufX, (const float4*)bufOut, N * (HID / 4));
    }

    out_kernel<<<(G + 7) / 8, 256, 0, stream>>>(hx, Wout, bout, (float*)d_out, G);
}
